// FastSpeech_24790551232885
// MI455X (gfx1250) — matrix operand order repaired and checked by reading, not compiled, not run on hardware
//
#include <hip/hip_runtime.h>
#include <hip/hip_bf16.h>

typedef __attribute__((ext_vector_type(16))) _Float16 v16h;
typedef __attribute__((ext_vector_type(8)))  _Float16 v8h;
typedef __attribute__((ext_vector_type(8)))  float    v8f;

namespace fs {
constexpr int B  = 16, T = 256, M = 2048, H = 512, FF = 2048, NH = 2, NL = 4;
constexpr int HD = H / NH;      // 256
constexpr int NM = 80;          // N_MELS
constexpr int NMP = 128;        // N_MELS padded to a full 2-tile-column block
constexpr int RE = B * T;       // 4096 encoder rows
constexpr int RD = B * M;       // 32768 decoder rows
constexpr float EPS = 1e-5f;
}

enum { GF_RELU = 1, GF_C32 = 2, GF_C16 = 4 };

// ---------------------------------------------------------------------------
// fp32 -> f16 conversion; _pad variant zero-fills [nsrc, ntotal)
// ---------------------------------------------------------------------------
__global__ void cvt_f32_to_f16(const float* __restrict__ s,
                               _Float16* __restrict__ d, int n) {
  int i = blockIdx.x * 256 + threadIdx.x;
  if (i < n) d[i] = (_Float16)s[i];
}

__global__ void cvt_f32_to_f16_pad(const float* __restrict__ s,
                                   _Float16* __restrict__ d, int nsrc, int ntotal) {
  int i = blockIdx.x * 256 + threadIdx.x;
  if (i < ntotal) d[i] = (i < nsrc) ? (_Float16)s[i] : (_Float16)0.0f;
}

// ---------------------------------------------------------------------------
// Embedding + positional embedding (indexed by BATCH, faithful to reference)
// grid = B*T blocks, 128 threads (float4 over H=512)
// ---------------------------------------------------------------------------
__global__ void embed_pos_kernel(const int* __restrict__ tokens,
                                 const float* __restrict__ embw,
                                 const float* __restrict__ pos,
                                 float* __restrict__ x32,
                                 _Float16* __restrict__ x16) {
  const int row = blockIdx.x;              // b*T + t
  const int b   = row / fs::T;
  const int tok = tokens[row];
  const int h   = threadIdx.x * 4;
  const float* e = embw + (size_t)tok * fs::H + h;
  const float* p = pos  + (size_t)b   * fs::H + h;
  const size_t o = (size_t)row * fs::H + h;
#pragma unroll
  for (int i = 0; i < 4; ++i) {
    float v = e[i] + p[i];
    x32[o + i] = v;
    x16[o + i] = (_Float16)v;
  }
}

// ---------------------------------------------------------------------------
// WMMA GEMM:  C[r,n] = sum_k A[r,k] * W[n,k] + bias[n]   (x @ W.T + b)
// A: [R,K] f16 row-major.  W: [>=gridDim.x*64, K] f16 row-major; caller pads W
// with zero rows so the K-loop needs NO column guards (exec stays all-ones,
// which WMMA requires anyway).
// Block = 256 threads = 8 waves; wave owns a 32x64 strip: 2 A fragments,
// 4 B fragments per 32-wide K step -> 8 v_wmma_f32_16x16x32_f16, B reused x2,
// A reused x4.  R must be a multiple of 256.
// ---------------------------------------------------------------------------
__global__ __launch_bounds__(256) void gemm_wmma_f16(
    const _Float16* __restrict__ A, const _Float16* __restrict__ W,
    const float* __restrict__ bias, float* C32, _Float16* C16,
    int K, int N, int flags) {
  const int lane = threadIdx.x & 31;
  const int wave = threadIdx.x >> 5;
  const int lm   = lane & 15;   // row (A) / col (B,C) inside a 16-wide tile
  const int hs   = lane >> 4;   // half-select per the 16-bit fragment layouts
  const int m0   = blockIdx.y * 256 + wave * 32;
  const int n0   = blockIdx.x * 64;

  const _Float16* arow0 = A + (size_t)(m0 + lm) * K;
  const _Float16* arow1 = A + (size_t)(m0 + 16 + lm) * K;

  v8f   acc[2][4] = {};
  int   ncol[4];
  float bcol[4];
#pragma unroll
  for (int j = 0; j < 4; ++j) {
    ncol[j] = n0 + j * 16 + lm;
    bcol[j] = (ncol[j] < N) ? bias[ncol[j]] : 0.0f;
  }

  for (int k = 0; k < K; k += 32) {
    // A fragment, 16x32 f16: lane<16 -> K {k..k+15},
    //                        lane>=16 -> K {k+16..k+31}
    union { v16h v; v8h h[2]; } a0, a1;
    a0.h[0] = *(const v8h*)(arow0 + k + (hs << 4));
    a0.h[1] = *(const v8h*)(arow0 + k + 8 + (hs << 4));
    a1.h[0] = *(const v8h*)(arow1 + k + (hs << 4));
    a1.h[1] = *(const v8h*)(arow1 + k + 8 + (hs << 4));
#pragma unroll
    for (int j = 0; j < 4; ++j) {
      // B fragment, 32x16 f16: lane = column, K chunk = k + (lane/16)*16
      v16h bm = *(const v16h*)(W + (size_t)ncol[j] * K + k + (hs << 4));
      acc[0][j] = __builtin_amdgcn_wmma_f32_16x16x32_f16(
          false, a0.v, false, bm, (short)0, acc[0][j], false, false);
      acc[1][j] = __builtin_amdgcn_wmma_f32_16x16x32_f16(
          false, a1.v, false, bm, (short)0, acc[1][j], false, false);
    }
  }

  // C/D layout: VGPR r -> M = (lane/16)*8 + r, N = n0 + j*16 + (lane%16)
#pragma unroll
  for (int p = 0; p < 2; ++p) {
    const int mbase = m0 + p * 16 + (hs << 3);
#pragma unroll
    for (int j = 0; j < 4; ++j) {
      if (ncol[j] >= N) continue;
#pragma unroll
      for (int r = 0; r < 8; ++r) {
        float v = acc[p][j][r] + bcol[j];
        if (flags & GF_RELU) v = fmaxf(v, 0.0f);
        size_t off = (size_t)(mbase + r) * N + ncol[j];
        if (flags & GF_C32) C32[off] = v;
        if (flags & GF_C16) C16[off] = (_Float16)v;
      }
    }
  }
}

// ---------------------------------------------------------------------------
// Attention over axis 0 (S = B = 16!), per (n, head) wave. ~0.5% of FLOPs.
// qkv: [S*N, 3H] f16 ;  out: [S*N, H] f16
// ---------------------------------------------------------------------------
__global__ __launch_bounds__(32) void attention_kernel(
    const _Float16* __restrict__ qkv, _Float16* __restrict__ out, int N) {
  const int n    = blockIdx.x >> 1;
  const int head = blockIdx.x & 1;
  const int lane = threadIdx.x;
  const int qo   = head * fs::HD;
  __shared__ float sc[16][17];

  for (int e = lane; e < 256; e += 32) {        // scores q.k / sqrt(256)
    const int s = e >> 4, t = e & 15;
    const _Float16* qp = qkv + ((size_t)s * N + n) * (3 * fs::H) + qo;
    const _Float16* kp = qkv + ((size_t)t * N + n) * (3 * fs::H) + fs::H + qo;
    float acc = 0.f;
    for (int d = 0; d < fs::HD; ++d) acc += (float)qp[d] * (float)kp[d];
    sc[s][t] = acc * 0.0625f;
  }
  __syncthreads();
  if (lane < 16) {                               // softmax over t
    float mx = -3.4e38f;
    for (int t = 0; t < 16; ++t) mx = fmaxf(mx, sc[lane][t]);
    float sum = 0.f;
    for (int t = 0; t < 16; ++t) { float e = expf(sc[lane][t] - mx); sc[lane][t] = e; sum += e; }
    float inv = 1.0f / sum;
    for (int t = 0; t < 16; ++t) sc[lane][t] *= inv;
  }
  __syncthreads();
  for (int e = lane; e < 16 * fs::HD; e += 32) { // o = att @ v
    const int s = e >> 8, d = e & 255;
    float acc = 0.f;
    for (int t = 0; t < 16; ++t)
      acc += sc[s][t] * (float)qkv[((size_t)t * N + n) * (3 * fs::H) + 2 * fs::H + qo + d];
    out[((size_t)s * N + n) * fs::H + qo + d] = (_Float16)acc;
  }
}

// ---------------------------------------------------------------------------
// y = LN(x + delta) * g + b ; writes fp32 + f16. One 256-thread block per row.
// ---------------------------------------------------------------------------
__global__ __launch_bounds__(256) void resid_ln_kernel(
    const float* xin, const float* delta, const float* __restrict__ g,
    const float* __restrict__ bb, float* yout, _Float16* __restrict__ y16) {
  const int row = blockIdx.x, tid = threadIdx.x;
  __shared__ float buf[256];
  const size_t base = (size_t)row * fs::H;
  float v0 = xin[base + tid]       + delta[base + tid];
  float v1 = xin[base + 256 + tid] + delta[base + 256 + tid];
  buf[tid] = v0 + v1;
  __syncthreads();
  for (int st = 128; st > 0; st >>= 1) {
    if (tid < st) buf[tid] += buf[tid + st];
    __syncthreads();
  }
  const float mu = buf[0] * (1.0f / fs::H);
  __syncthreads();
  const float d0 = v0 - mu, d1 = v1 - mu;
  buf[tid] = d0 * d0 + d1 * d1;
  __syncthreads();
  for (int st = 128; st > 0; st >>= 1) {
    if (tid < st) buf[tid] += buf[tid + st];
    __syncthreads();
  }
  const float rstd = rsqrtf(buf[0] * (1.0f / fs::H) + fs::EPS);
  const float o0 = d0 * rstd * g[tid] + bb[tid];
  const float o1 = d1 * rstd * g[256 + tid] + bb[256 + tid];
  yout[base + tid]       = o0;  y16[base + tid]       = (_Float16)o0;
  yout[base + 256 + tid] = o1;  y16[base + 256 + tid] = (_Float16)o1;
}

// ---------------------------------------------------------------------------
// Length regulator index table: clipped cumsum of normalized durations,
// searchsorted(side='right'). grid = B, 256 threads (T = 256).
// ---------------------------------------------------------------------------
__global__ __launch_bounds__(256) void lr_index_kernel(
    const float* __restrict__ align, const int* __restrict__ mgn,
    int* __restrict__ idx) {
  const int b = blockIdx.x, tid = threadIdx.x;
  __shared__ float c[256];
  __shared__ float red[256];
  const float a = align[b * fs::T + tid];
  red[tid] = a;
  __syncthreads();
  for (int st = 128; st > 0; st >>= 1) {
    if (tid < st) red[tid] += red[tid + st];
    __syncthreads();
  }
  const float mg    = (float)mgn[b];
  const float scale = mg / red[0];
  c[tid] = a * scale;
  __syncthreads();
  for (int off = 1; off < 256; off <<= 1) {      // Hillis-Steele inclusive scan
    float t = (tid >= off) ? c[tid - off] : 0.0f;
    __syncthreads();
    c[tid] += t;
    __syncthreads();
  }
  c[tid] = fminf(c[tid], mg);
  __syncthreads();
  for (int m = tid; m < fs::M; m += 256) {
    const float f = (float)m;
    int lo = 0, hi = fs::T;
    while (lo < hi) {                            // first i with c[i] > f
      int mid = (lo + hi) >> 1;
      if (c[mid] > f) hi = mid; else lo = mid + 1;
    }
    if (lo > fs::T - 1) lo = fs::T - 1;
    idx[b * fs::M + m] = (f < mg) ? lo : -1;
  }
}

// gather + second pos-emb add. grid = B*M blocks, 128 threads.
__global__ void lr_gather_kernel(const float* __restrict__ xe,
                                 const int* __restrict__ idx,
                                 const float* __restrict__ pos,
                                 float* __restrict__ x32,
                                 _Float16* __restrict__ x16) {
  const int row = blockIdx.x;            // b*M + m
  const int b   = row / fs::M;
  const int id  = idx[row];
  const int h   = threadIdx.x * 4;
  const float* p   = pos + (size_t)b * fs::H + h;
  const float* src = xe + ((size_t)b * fs::T + (id >= 0 ? id : 0)) * fs::H + h;
  const size_t o = (size_t)row * fs::H + h;
#pragma unroll
  for (int i = 0; i < 4; ++i) {
    float v = (id >= 0 ? src[i] : 0.0f) + p[i];
    x32[o + i] = v;
    x16[o + i] = (_Float16)v;
  }
}

// ---------------------------------------------------------------------------
extern "C" void kernel_launch(void* const* d_in, const int* in_sizes, int n_in,
                              void* d_out, int out_size, void* d_ws, size_t ws_size,
                              hipStream_t stream) {
  using namespace fs;
  (void)in_sizes; (void)n_in; (void)out_size; (void)ws_size;

  const int*   tokens = (const int*)d_in[0];
  const float* align  = (const float*)d_in[1];
  const int*   mgn    = (const int*)d_in[2];
  const float* embw   = (const float*)d_in[4];
  const float* pos    = (const float*)d_in[5];
  const float* qkvW32 = (const float*)d_in[6];
  const float* qkvB   = (const float*)d_in[7];
  const float* outW32 = (const float*)d_in[8];
  const float* outB   = (const float*)d_in[9];
  const float* ln1g   = (const float*)d_in[10];
  const float* ln1b   = (const float*)d_in[11];
  const float* ln2g   = (const float*)d_in[12];
  const float* ln2b   = (const float*)d_in[13];
  const float* f1W32  = (const float*)d_in[14];
  const float* f1B    = (const float*)d_in[15];
  const float* f2W32  = (const float*)d_in[16];
  const float* f2B    = (const float*)d_in[17];
  const float* linW32 = (const float*)d_in[18];
  const float* linB   = (const float*)d_in[19];

  char* cur = (char*)d_ws;
  auto alloc = [&](size_t bytes) -> void* {
    void* r = (void*)cur;
    cur += (bytes + 255) & ~(size_t)255;
    return r;
  };

  _Float16* wq16  = (_Float16*)alloc((size_t)NL * 3 * H * H * 2);
  _Float16* wo16  = (_Float16*)alloc((size_t)NL * H * H * 2);
  _Float16* wf1   = (_Float16*)alloc((size_t)NL * FF * H * 2);
  _Float16* wf2   = (_Float16*)alloc((size_t)NL * H * FF * 2);
  _Float16* wl16  = (_Float16*)alloc((size_t)NMP * H * 2);   // zero-padded 80->128 rows
  float*    xe32  = (float*)   alloc((size_t)RE * H * 4);
  _Float16* xe16  = (_Float16*)alloc((size_t)RE * H * 2);
  float*    xd32  = (float*)   alloc((size_t)RD * H * 4);
  _Float16* xd16  = (_Float16*)alloc((size_t)RD * H * 2);
  _Float16* attno = (_Float16*)alloc((size_t)RD * H * 2);
  float*    delta = (float*)   alloc((size_t)RD * H * 4);
  _Float16* bigR  = (_Float16*)alloc((size_t)RD * FF * 2); // qkv16/h16 alias (disjoint lifetimes)
  int*      idxb  = (int*)     alloc((size_t)RD * 4);
  _Float16* qkv16 = bigR;
  _Float16* h16   = bigR;

  auto cvt = [&](const float* s, _Float16* d, size_t n) {
    cvt_f32_to_f16<<<dim3((unsigned)((n + 255) / 256)), dim3(256), 0, stream>>>(s, d, (int)n);
  };
  cvt(qkvW32, wq16, (size_t)NL * 3 * H * H);
  cvt(outW32, wo16, (size_t)NL * H * H);
  cvt(f1W32,  wf1,  (size_t)NL * FF * H);
  cvt(f2W32,  wf2,  (size_t)NL * H * FF);
  cvt_f32_to_f16_pad<<<dim3((NMP * H + 255) / 256), dim3(256), 0, stream>>>(
      linW32, wl16, NM * H, NMP * H);

  embed_pos_kernel<<<RE, 128, 0, stream>>>(tokens, embw, pos, xe32, xe16);

  auto run_layer = [&](int i, float* x32, _Float16* x16, int R, int Nn) {
    gemm_wmma_f16<<<dim3(3 * H / 64, R / 256), 256, 0, stream>>>(
        x16, wq16 + (size_t)i * 3 * H * H, qkvB + i * 3 * H,
        nullptr, qkv16, H, 3 * H, GF_C16);
    attention_kernel<<<Nn * NH, 32, 0, stream>>>(qkv16, attno, Nn);
    gemm_wmma_f16<<<dim3(H / 64, R / 256), 256, 0, stream>>>(
        attno, wo16 + (size_t)i * H * H, outB + i * H,
        delta, nullptr, H, H, GF_C32);
    resid_ln_kernel<<<R, 256, 0, stream>>>(x32, delta, ln1g + i * H, ln1b + i * H, x32, x16);
    gemm_wmma_f16<<<dim3(FF / 64, R / 256), 256, 0, stream>>>(
        x16, wf1 + (size_t)i * FF * H, f1B + i * FF,
        nullptr, h16, H, FF, GF_RELU | GF_C16);
    gemm_wmma_f16<<<dim3(H / 64, R / 256), 256, 0, stream>>>(
        h16, wf2 + (size_t)i * H * FF, f2B + i * H,
        delta, nullptr, FF, H, GF_C32);
    resid_ln_kernel<<<R, 256, 0, stream>>>(x32, delta, ln2g + i * H, ln2b + i * H, x32, x16);
  };

  // Encoder (attention seq = B = 16, N = T)
  run_layer(0, xe32, xe16, RE, T);
  run_layer(1, xe32, xe16, RE, T);

  // Length regulator + second positional add
  lr_index_kernel<<<B, 256, 0, stream>>>(align, mgn, idxb);
  lr_gather_kernel<<<RD, 128, 0, stream>>>(xe32, idxb, pos, xd32, xd16);

  // Decoder (N = M)
  run_layer(2, xd32, xd16, RD, M);
  run_layer(3, xd32, xd16, RD, M);

  // Final projection H -> 80 mels (weights zero-padded to 128 cols),
  // straight into d_out (fp32)
  gemm_wmma_f16<<<dim3(NMP / 64, RD / 256), 256, 0, stream>>>(
      xd16, wl16, linB, (float*)d_out, nullptr, H, NM, GF_C32);
}
